// MultiHeadAttention_48936857371058
// MI455X (gfx1250) — compile-verified, matrix operand order checked by reading, not run on hardware
//
#include <hip/hip_runtime.h>
#include <math.h>

typedef __attribute__((ext_vector_type(16))) _Float16 v16h;
typedef __attribute__((ext_vector_type(8)))  _Float16 v8h;
typedef __attribute__((ext_vector_type(8)))  float    v8f;
typedef __attribute__((ext_vector_type(4)))  float    v4f;
typedef __attribute__((ext_vector_type(4)))  int      v4i;

#define D_MODEL 512
#define SEQ     2048
#define HEADS   8
#define DK      64
#define BATCH   8

union Frag16U { v16h v; v8h h[2]; };

// Fragment layout model (CDNA5 ISA 7.12.2, 16-bit A 16x32):
//   lane<16 : row/col = lane,    K = {k0..k0+7, k0+16..k0+23}
//   lane>=16: row/col = lane-16, K = {k0+8..k0+15, k0+24..k0+31}
// B (32x16) mirrors A (per-lane column). Same contiguous per-lane load pattern.
__device__ __forceinline__ v16h load_frag16(const _Float16* p, size_t stride, int r, int k0) {
  const int lane = threadIdx.x & 31;
  const _Float16* q = p + (size_t)(r + (lane & 15)) * stride + (size_t)(k0 + ((lane >> 4) << 3));
  Frag16U u;
  u.h[0] = *(const v8h*)q;
  u.h[1] = *(const v8h*)(q + 16);
  return u.v;
}

__device__ __forceinline__ v8f wmma_f16(v16h a, v16h b, v8f c) {
  return __builtin_amdgcn_wmma_f32_16x16x32_f16(false, a, false, b, (short)0, c, false, false);
}

__device__ __forceinline__ v8f zero8() {
  v8f z = {0.f, 0.f, 0.f, 0.f, 0.f, 0.f, 0.f, 0.f};
  return z;
}

// Bandwidth-bound fp32 -> fp16 conversion, 8 elements / thread.
__global__ void cvt_f16_kernel(const float* __restrict__ src, _Float16* __restrict__ dst, int n8) {
  const int i = (int)(blockIdx.x * blockDim.x + threadIdx.x);
  if (i >= n8) return;
  const size_t base = (size_t)i * 8;
  v4f a = *(const v4f*)(src + base);
  v4f b = *(const v4f*)(src + base + 4);
  v8h w;
#pragma unroll
  for (int j = 0; j < 4; ++j) {
    w[j]     = (_Float16)a[j];
    w[j + 4] = (_Float16)b[j];
  }
  *(v8h*)(dst + base) = w;
}

// T = (X @ W^T)^T computed as W @ X^T; stored head-split f16 [b,h,s,dk].
// Wave = one 16-row (s) x 64-col (n) tile; strip (= head) = wave&7.
__global__ void proj_qk_kernel(const _Float16* __restrict__ X, const _Float16* __restrict__ W,
                               const float* __restrict__ bias, _Float16* __restrict__ out) {
  const int lane = threadIdx.x & 31;
  const int wave = (int)((blockIdx.x * blockDim.x + threadIdx.x) >> 5);
  const int strip = wave & 7;
  const int m0 = (wave >> 3) << 4;
  v8f acc[4];
#pragma unroll
  for (int t = 0; t < 4; ++t) acc[t] = zero8();

  for (int k0 = 0; k0 < D_MODEL; k0 += 32) {
    v16h bf = load_frag16(X, D_MODEL, m0, k0);          // B: columns = X rows
#pragma unroll
    for (int t = 0; t < 4; ++t) {
      v16h af = load_frag16(W, D_MODEL, strip * 64 + t * 16, k0);  // A: W rows
      acc[t] = wmma_f16(af, bf, acc[t]);
    }
  }

  const int m = m0 + (lane & 15);          // global token row = b*SEQ + s
  const int b = m >> 11, s = m & (SEQ - 1);
  const int up8 = (lane >> 4) << 3;
#pragma unroll
  for (int t = 0; t < 4; ++t) {
    const int nb = strip * 64 + t * 16 + up8;    // h = strip, d = t*16+up8
    v4f bb0 = *(const v4f*)(bias + nb);
    v4f bb1 = *(const v4f*)(bias + nb + 4);
    v8h w;
#pragma unroll
    for (int i = 0; i < 4; ++i) {
      w[i]     = (_Float16)(acc[t][i] + bb0[i]);
      w[i + 4] = (_Float16)(acc[t][i + 4] + bb1[i]);
    }
    size_t off = (((size_t)(b * HEADS + strip) * SEQ) + (size_t)s) * DK + (size_t)(t * 16 + up8);
    *(v8h*)(out + off) = w;
  }
}

// C = X @ W^T, stored TRANSPOSED head-split f16 [b,h,dk,s] (for V^T A-fragments).
__global__ void proj_v_kernel(const _Float16* __restrict__ X, const _Float16* __restrict__ W,
                              const float* __restrict__ bias, _Float16* __restrict__ outT) {
  const int lane = threadIdx.x & 31;
  const int wave = (int)((blockIdx.x * blockDim.x + threadIdx.x) >> 5);
  const int strip = wave & 7;
  const int m0 = (wave >> 3) << 4;
  v8f acc[4];
#pragma unroll
  for (int t = 0; t < 4; ++t) acc[t] = zero8();

  for (int k0 = 0; k0 < D_MODEL; k0 += 32) {
    v16h af = load_frag16(X, D_MODEL, m0, k0);          // A: X rows (s)
#pragma unroll
    for (int t = 0; t < 4; ++t) {
      v16h bf = load_frag16(W, D_MODEL, strip * 64 + t * 16, k0);  // B: cols = W rows
      acc[t] = wmma_f16(af, bf, acc[t]);
    }
  }

  const int b = m0 >> 11;
  const int up8 = (lane >> 4) << 3;
  const int s0 = (m0 & (SEQ - 1)) + up8;   // 8 consecutive s per lane
#pragma unroll
  for (int t = 0; t < 4; ++t) {
    const int d = t * 16 + (lane & 15);
    const float bv = bias[strip * 64 + d];
    v8h w;
#pragma unroll
    for (int i = 0; i < 8; ++i) w[i] = (_Float16)(acc[t][i] + bv);
    size_t off = (((size_t)(b * HEADS + strip) * DK) + (size_t)d) * SEQ + (size_t)s0;
    *(v8h*)(outT + off) = w;
  }
}

// Flash attention, one wave per (b,h,16-query tile), 32-key steps.
// Scores computed transposed (keys=M, queries=N) so softmax is per-lane +
// one shfl_xor(16); P^T in C-layout IS the B-fragment for O^T = V^T @ P^T.
__global__ void attn_kernel(const _Float16* __restrict__ Qp, const _Float16* __restrict__ Kp,
                            const _Float16* __restrict__ Vt, const int* __restrict__ mask,
                            _Float16* __restrict__ attn) {
  const int lane = threadIdx.x & 31;
  const int wave = (int)((blockIdx.x * blockDim.x + threadIdx.x) >> 5);
  const int bh = wave >> 7;              // 0..63
  const int qbase = (wave & 127) << 4;
  const int up8 = (lane >> 4) << 3;
  const int qrow = qbase + (lane & 15);

  const _Float16* Qbh = Qp + (size_t)bh * SEQ * DK;
  const _Float16* Kbh = Kp + (size_t)bh * SEQ * DK;
  const _Float16* Vbh = Vt + (size_t)bh * DK * SEQ;

  const v16h qf0 = load_frag16(Qbh, DK, qbase, 0);
  const v16h qf1 = load_frag16(Qbh, DK, qbase, 32);

  v8f o[4];
#pragma unroll
  for (int t = 0; t < 4; ++t) o[t] = zero8();
  float mrun = -1e30f, lrun = 0.f;

  for (int kb = 0; kb < SEQ; kb += 32) {
    if (kb + 32 < SEQ) {   // cover L2 latency of the streaming K/V walk
      __builtin_prefetch(Kbh + (size_t)(kb + 32) * DK + (size_t)(lane << 6), 0, 1);
      __builtin_prefetch(Vbh + (size_t)(lane << 1) * SEQ + (size_t)(kb + 32), 0, 1);
    }
    // S^T tiles: keys kb..kb+15 and kb+16..kb+31 as M, queries as N, K=dk=64.
    v8f s0 = zero8(), s1 = zero8();
    s0 = wmma_f16(load_frag16(Kbh, DK, kb, 0),       qf0, s0);
    s0 = wmma_f16(load_frag16(Kbh, DK, kb, 32),      qf1, s0);
    s1 = wmma_f16(load_frag16(Kbh, DK, kb + 16, 0),  qf0, s1);
    s1 = wmma_f16(load_frag16(Kbh, DK, kb + 16, 32), qf1, s1);

    // mask row for this lane's query; this lane's key columns.
    const int* mrow = mask + (size_t)qrow * SEQ + (size_t)(kb + up8);
    v4i ma0 = *(const v4i*)(mrow);
    v4i ma1 = *(const v4i*)(mrow + 4);
    v4i mb0 = *(const v4i*)(mrow + 16);
    v4i mb1 = *(const v4i*)(mrow + 20);

    float sv0[8], sv1[8];
#pragma unroll
    for (int i = 0; i < 4; ++i) {
      sv0[i]     = ma0[i] ? s0[i] * 0.125f     : -1e30f;
      sv0[i + 4] = ma1[i] ? s0[i + 4] * 0.125f : -1e30f;
      sv1[i]     = mb0[i] ? s1[i] * 0.125f     : -1e30f;
      sv1[i + 4] = mb1[i] ? s1[i + 4] * 0.125f : -1e30f;
    }
    float mx = -1e30f;
#pragma unroll
    for (int i = 0; i < 8; ++i) mx = fmaxf(mx, fmaxf(sv0[i], sv1[i]));
    mx = fmaxf(mx, __shfl_xor(mx, 16, 32));   // combine lane-pair (same query)
    const float mnew = fmaxf(mrun, mx);
    const float alpha = __expf(mrun - mnew);

    v16h p;
    float lsum = 0.f;
#pragma unroll
    for (int i = 0; i < 8; ++i) {
      float e0 = __expf(sv0[i] - mnew);   // keys kb+up8+i
      float e1 = __expf(sv1[i] - mnew);   // keys kb+16+up8+i
      lsum += e0 + e1;
      p[i] = (_Float16)e0;                // B-frag halves: K{0..7|8..15}, K{16..23|24..31}
      p[i + 8] = (_Float16)e1;
    }
    lsum += __shfl_xor(lsum, 16, 32);
    lrun = lrun * alpha + lsum;
    mrun = mnew;

#pragma unroll
    for (int t = 0; t < 4; ++t) {
#pragma unroll
      for (int i = 0; i < 8; ++i) o[t][i] *= alpha;
      // A = V^T rows (dk = t*16.. ), K = 32 keys at kb
      o[t] = wmma_f16(load_frag16(Vbh, SEQ, t * 16, kb), p, o[t]);
    }
  }

  const float inv = 1.0f / lrun;
  const int b = bh >> 3, h = bh & 7;
#pragma unroll
  for (int t = 0; t < 4; ++t) {
    v8h w;
#pragma unroll
    for (int i = 0; i < 8; ++i) w[i] = (_Float16)(o[t][i] * inv);
    size_t off = ((size_t)(b * SEQ + qrow)) * D_MODEL + (size_t)(h * DK + t * 16 + up8);
    *(v8h*)(attn + off) = w;
  }
}

// Out = Attn @ Wo^T + bo, fp32, computed transposed (A = Wo rows, B = Attn rows)
// so each lane stores 8 consecutive fp32 columns of its row.
__global__ void proj_out_kernel(const _Float16* __restrict__ Xh, const _Float16* __restrict__ W,
                                const float* __restrict__ bias, float* __restrict__ out) {
  const int lane = threadIdx.x & 31;
  const int wave = (int)((blockIdx.x * blockDim.x + threadIdx.x) >> 5);
  const int strip = wave & 7;
  const int m0 = (wave >> 3) << 4;
  v8f acc[4];
#pragma unroll
  for (int t = 0; t < 4; ++t) acc[t] = zero8();

  for (int k0 = 0; k0 < D_MODEL; k0 += 32) {
    v16h bf = load_frag16(Xh, D_MODEL, m0, k0);
#pragma unroll
    for (int t = 0; t < 4; ++t) {
      v16h af = load_frag16(W, D_MODEL, strip * 64 + t * 16, k0);
      acc[t] = wmma_f16(af, bf, acc[t]);
    }
  }

  const int m = m0 + (lane & 15);
  const int up8 = (lane >> 4) << 3;
#pragma unroll
  for (int t = 0; t < 4; ++t) {
    const int nb = strip * 64 + t * 16 + up8;
    v4f bb0 = *(const v4f*)(bias + nb);
    v4f bb1 = *(const v4f*)(bias + nb + 4);
    v4f r0, r1;
#pragma unroll
    for (int i = 0; i < 4; ++i) {
      r0[i] = acc[t][i] + bb0[i];
      r1[i] = acc[t][i + 4] + bb1[i];
    }
    float* dst = out + (size_t)m * D_MODEL + (size_t)nb;
    *(v4f*)(dst) = r0;
    *(v4f*)(dst + 4) = r1;
  }
}

extern "C" void kernel_launch(void* const* d_in, const int* in_sizes, int n_in,
                              void* d_out, int out_size, void* d_ws, size_t ws_size,
                              hipStream_t stream) {
  const float* Q    = (const float*)d_in[0];
  const float* K    = (const float*)d_in[1];
  const float* V    = (const float*)d_in[2];
  const int*   mask = (const int*)d_in[3];
  const float* Wq = (const float*)d_in[4];  const float* bq = (const float*)d_in[5];
  const float* Wk = (const float*)d_in[6];  const float* bk = (const float*)d_in[7];
  const float* Wv = (const float*)d_in[8];  const float* bv = (const float*)d_in[9];
  const float* Wo = (const float*)d_in[10]; const float* bo = (const float*)d_in[11];
  float* out = (float*)d_out;

  _Float16* ws = (_Float16*)d_ws;
  const size_t HSZ = (size_t)BATCH * HEADS * SEQ * DK;   // 8.39M f16 elements
  _Float16* Qp = ws;              // [b,h,s,dk]
  _Float16* Kp = ws + HSZ;        // [b,h,s,dk]
  _Float16* Vt = ws + 2 * HSZ;    // [b,h,dk,s]
  _Float16* At = ws + 3 * HSZ;    // [b,s,d_model]
  _Float16* Xh = ws + 4 * HSZ;    // f16 staging for Q/K/V (reused serially)
  _Float16* Wh = ws + 5 * HSZ;    // f16 staging for Wq/Wk/Wv/Wo (reused serially)

  const dim3 blk(256);
  const dim3 grd(1024);           // 8192 waves: 8 n-strips x 1024 m-tiles (or 64 bh x 128 q-tiles)
  const int nX8 = (int)((size_t)BATCH * SEQ * D_MODEL / 8);  // 1048576
  const int nW8 = D_MODEL * D_MODEL / 8;                     // 32768
  const dim3 grdX(nX8 / 256);     // 4096
  const dim3 grdW(nW8 / 256);     // 128

  // Q projection
  cvt_f16_kernel<<<grdX, blk, 0, stream>>>(Q, Xh, nX8);
  cvt_f16_kernel<<<grdW, blk, 0, stream>>>(Wq, Wh, nW8);
  proj_qk_kernel<<<grd, blk, 0, stream>>>(Xh, Wh, bq, Qp);
  // K projection
  cvt_f16_kernel<<<grdX, blk, 0, stream>>>(K, Xh, nX8);
  cvt_f16_kernel<<<grdW, blk, 0, stream>>>(Wk, Wh, nW8);
  proj_qk_kernel<<<grd, blk, 0, stream>>>(Xh, Wh, bk, Kp);
  // V projection (stored transposed per head)
  cvt_f16_kernel<<<grdX, blk, 0, stream>>>(V, Xh, nX8);
  cvt_f16_kernel<<<grdW, blk, 0, stream>>>(Wv, Wh, nW8);
  proj_v_kernel <<<grd, blk, 0, stream>>>(Xh, Wh, bv, Vt);
  // Attention
  attn_kernel   <<<grd, blk, 0, stream>>>(Qp, Kp, Vt, mask, At);
  // Output projection
  cvt_f16_kernel<<<grdW, blk, 0, stream>>>(Wo, Wh, nW8);
  proj_out_kernel<<<grd, blk, 0, stream>>>(At, Wh, bo, out);
}
